// EnhancementBlock_68959994904805
// MI455X (gfx1250) — compile-verified, matrix operand order checked by reading, not run on hardware
//
#include <hip/hip_runtime.h>

typedef __attribute__((ext_vector_type(16))) __bf16 bf16x16;
typedef __attribute__((ext_vector_type(8)))  float  f32x8;

union Frag { bf16x16 v; uint4 q[2]; };

// ---------------------------------------------------------------------------
// K0a: convert w_dcn (128 x 1152 f32, already [o][i*9+k] order) to bf16
// ---------------------------------------------------------------------------
__global__ __launch_bounds__(256) void k0_convert_wdcn(const float* __restrict__ w_dcn,
                                                       __bf16* __restrict__ wA, int n) {
    int i = blockIdx.x * 256 + threadIdx.x;
    if (i < n) wA[i] = (__bf16)w_dcn[i];
}

// ---------------------------------------------------------------------------
// K0b: convert w_off (27 x 1152 f32, flat order matches [oc][ci*9+t]) to bf16,
//      zero-padded to 32 output rows for WMMA M-tiling.
// ---------------------------------------------------------------------------
__global__ __launch_bounds__(256) void k0_convert_woff(const float* __restrict__ w_off,
                                                       __bf16* __restrict__ wOff) {
    int i = blockIdx.x * 256 + threadIdx.x;           // 0 .. 32*1152-1
    if (i < 32 * 1152) wOff[i] = (i < 27 * 1152) ? (__bf16)w_off[i] : (__bf16)0.f;
}

// ---------------------------------------------------------------------------
// K1: offset conv as bf16 WMMA im2col GEMM. M=32(pad of 27), K=1152, N=32768.
//     Epilogue fuses base-grid add (dy/dx) and sigmoid(mask).
//     Grid: 256 blocks (128-position N tile), 8 waves = 8 N-blocks of 16.
// ---------------------------------------------------------------------------
__global__ __launch_bounds__(256) void k1_offset_wmma(const float* __restrict__ x,
                                                      const __bf16* __restrict__ wOff,
                                                      const float* __restrict__ b_off,
                                                      float* __restrict__ ysb,
                                                      float* __restrict__ xsb,
                                                      float* __restrict__ mb) {
    __shared__ __attribute__((aligned(16))) __bf16 lds_a[32][40];    // A: 32 x 32 (+pad)
    __shared__ __attribute__((aligned(16))) __bf16 lds_b[128][40];   // B: N-major, 128 x 32 (+pad)

    const int tid  = threadIdx.x;
    const int lane = tid & 31;
    const int wid  = tid >> 5;          // wave_n: 0..7, N-block of 16
    const int p0   = blockIdx.x * 128;
    const int b    = p0 >> 12;
    const int hw0  = p0 & 4095;

    f32x8 acc[2] = {};
    const float* __restrict__ xb = x + (size_t)b * 128 * 4096;

    for (int kc = 0; kc < 36; ++kc) {
        const int idx0 = kc * 32;
        __syncthreads();
        // A tile: 32 rows x 32 cols = 128 x uint4 chunks
        if (tid < 128) {
            int r = tid >> 2, cb = (tid & 3) * 8;
            *(uint4*)&lds_a[r][cb] = *(const uint4*)(wOff + r * 1152 + idx0 + cb);
        }
        if (kc + 1 < 36) __builtin_prefetch(wOff + (idx0 + 32), 0, 1);
        // B tile: im2col gather, stored N-major
        for (int e = tid; e < 4096; e += 256) {
            int kk = e >> 7, j = e & 127;
            int idx = idx0 + kk;
            int ci = idx / 9;
            int t  = idx - ci * 9;
            int ky = t / 3, kx = t - ky * 3;
            int hw = hw0 + j;
            int iy = (hw >> 6) + ky - 1;
            int ix = (hw & 63) + kx - 1;
            float v = (iy >= 0 && iy < 64 && ix >= 0 && ix < 64)
                          ? xb[ci * 4096 + iy * 64 + ix] : 0.f;
            lds_b[j][kk] = (__bf16)v;
        }
        __syncthreads();

        const int kb = (lane >> 4) << 3;       // 0 or 8
        Frag afr[2];
#pragma unroll
        for (int mt = 0; mt < 2; ++mt) {
            int row = mt * 16 + (lane & 15);
            afr[mt].q[0] = *(const uint4*)&lds_a[row][kb];
            afr[mt].q[1] = *(const uint4*)&lds_a[row][kb + 16];
        }
        Frag bfr;
        {
            int col = wid * 16 + (lane & 15);
            bfr.q[0] = *(const uint4*)&lds_b[col][kb];
            bfr.q[1] = *(const uint4*)&lds_b[col][kb + 16];
        }
#pragma unroll
        for (int mt = 0; mt < 2; ++mt)
            acc[mt] = __builtin_amdgcn_wmma_f32_16x16x32_bf16(
                false, afr[mt].v, false, bfr.v, (short)0, acc[mt], false, false);
    }

    // epilogue: apply offset/mask transform per output channel
    int n  = wid * 16 + (lane & 15);
    int hw = hw0 + n;
    int h = hw >> 6, w = hw & 63;
#pragma unroll
    for (int mt = 0; mt < 2; ++mt)
#pragma unroll
        for (int v = 0; v < 8; ++v) {
            int oc = mt * 16 + ((lane >> 4) << 3) + v;
            if (oc >= 27) continue;
            float val = acc[mt][v] + b_off[oc];
            if (oc < 18) {
                int k = oc >> 1;
                if ((oc & 1) == 0)
                    ysb[(b * 9 + k) * 4096 + hw] = (float)(h + (k / 3) - 1) + val;
                else
                    xsb[(b * 9 + k) * 4096 + hw] = (float)(w + (k % 3) - 1) + val;
            } else {
                int k = oc - 18;
                mb[(b * 9 + k) * 4096 + hw] = 1.f / (1.f + expf(-val));
            }
        }
}

// ---------------------------------------------------------------------------
// K2: DCN GEMM via bf16 WMMA with fused bilinear sampling.
//     M=128, K=1152, N=32768. 8 waves (2 M-blocks x 4 N-blocks), 4x2 WMMAs.
//     B tile stored N-major so fragments load as 2x ds_load_b128.
// ---------------------------------------------------------------------------
__global__ __launch_bounds__(256) void k2_dcn_wmma(const float* __restrict__ x,
                                                   const __bf16* __restrict__ wA,
                                                   const float* __restrict__ ysb,
                                                   const float* __restrict__ xsb,
                                                   const float* __restrict__ mb,
                                                   const float* __restrict__ b_dcn,
                                                   float* __restrict__ y_pre) {
    __shared__ __attribute__((aligned(16))) __bf16 lds_a[128][40];   // A: M-major
    __shared__ __attribute__((aligned(16))) __bf16 lds_b[128][40];   // B: N-major
    __shared__ float lds_ys[9][128], lds_xs[9][128], lds_m[9][128];

    const int tid    = threadIdx.x;
    const int lane   = tid & 31;
    const int wid    = tid >> 5;
    const int wave_m = wid & 1;         // 0..1  (64 rows each)
    const int wave_n = wid >> 1;        // 0..3  (32 cols each)
    const int p0  = blockIdx.x * 128;
    const int b   = p0 >> 12;
    const int hw0 = p0 & 4095;

    for (int e = tid; e < 9 * 128; e += 256) {
        int t = e >> 7, j = e & 127;
        int idx = (b * 9 + t) * 4096 + hw0 + j;
        lds_ys[t][j] = ysb[idx];
        lds_xs[t][j] = xsb[idx];
        lds_m[t][j]  = mb[idx];
    }

    f32x8 acc[4][2] = {};
    const float* __restrict__ xb = x + (size_t)b * 128 * 4096;

    for (int kc = 0; kc < 36; ++kc) {
        const int idx0 = kc * 32;
        __syncthreads();
        // A tile (128x32 bf16) as 16B chunks
        for (int e = tid; e < 512; e += 256) {
            int r = e >> 2, cb = (e & 3) * 8;
            *(uint4*)&lds_a[r][cb] = *(const uint4*)(wA + r * 1152 + idx0 + cb);
        }
        if (kc + 1 < 36) __builtin_prefetch(wA + (idx0 + 32), 0, 1);
        // B tile: fused bilinear gather, stored N-major
        for (int e = tid; e < 4096; e += 256) {
            int kk = e >> 7, j = e & 127;
            int idx = idx0 + kk;
            int ci = idx / 9;
            int t  = idx - ci * 9;
            float yv = lds_ys[t][j], xv = lds_xs[t][j], mk = lds_m[t][j];
            float y0f = floorf(yv), x0f = floorf(xv);
            float wy = yv - y0f, wx = xv - x0f;
            int y0 = (int)y0f, x0 = (int)x0f;
            const float* img = xb + ci * 4096;
            float v00 = (y0 >= 0   && y0 < 64   && x0 >= 0   && x0 < 64)   ? img[y0 * 64 + x0]           : 0.f;
            float v01 = (y0 >= 0   && y0 < 64   && x0+1 >= 0 && x0+1 < 64) ? img[y0 * 64 + x0 + 1]       : 0.f;
            float v10 = (y0+1 >= 0 && y0+1 < 64 && x0 >= 0   && x0 < 64)   ? img[(y0 + 1) * 64 + x0]     : 0.f;
            float v11 = (y0+1 >= 0 && y0+1 < 64 && x0+1 >= 0 && x0+1 < 64) ? img[(y0 + 1) * 64 + x0 + 1] : 0.f;
            float v = v00 * (1.f - wy) * (1.f - wx) + v01 * (1.f - wy) * wx +
                      v10 * wy * (1.f - wx)         + v11 * wy * wx;
            lds_b[j][kk] = (__bf16)(v * mk);
        }
        __syncthreads();

        const int kb = (lane >> 4) << 3;     // 0 or 8
        Frag afr[4];
#pragma unroll
        for (int mt = 0; mt < 4; ++mt) {
            int row = wave_m * 64 + mt * 16 + (lane & 15);
            afr[mt].q[0] = *(const uint4*)&lds_a[row][kb];
            afr[mt].q[1] = *(const uint4*)&lds_a[row][kb + 16];
        }
        Frag bfr[2];
#pragma unroll
        for (int nt = 0; nt < 2; ++nt) {
            int col = wave_n * 32 + nt * 16 + (lane & 15);
            bfr[nt].q[0] = *(const uint4*)&lds_b[col][kb];
            bfr[nt].q[1] = *(const uint4*)&lds_b[col][kb + 16];
        }
#pragma unroll
        for (int mt = 0; mt < 4; ++mt)
#pragma unroll
            for (int nt = 0; nt < 2; ++nt)
                acc[mt][nt] = __builtin_amdgcn_wmma_f32_16x16x32_bf16(
                    false, afr[mt].v, false, bfr[nt].v, (short)0, acc[mt][nt], false, false);
    }

    // epilogue: D layout -> NCHW f32 (+ conv bias)
#pragma unroll
    for (int mt = 0; mt < 4; ++mt)
        for (int nt = 0; nt < 2; ++nt) {
            int n  = wave_n * 32 + nt * 16 + (lane & 15);
            int hw = hw0 + n;
#pragma unroll
            for (int v = 0; v < 8; ++v) {
                int o = wave_m * 64 + mt * 16 + ((lane >> 4) << 3) + v;
                y_pre[((size_t)(b * 128 + o)) * 4096 + hw] = acc[mt][nt][v] + b_dcn[o];
            }
        }
}

// ---------------------------------------------------------------------------
// K3: per-channel BN statistics (mean, rsqrt(var+eps)) over B*H*W
// ---------------------------------------------------------------------------
__global__ __launch_bounds__(256) void k3_bn_stats(const float* __restrict__ y_pre,
                                                   float* __restrict__ mu,
                                                   float* __restrict__ rsig) {
    __shared__ float ssum[256], ssq[256];
    int c = blockIdx.x, tid = threadIdx.x;
    float s = 0.f, q = 0.f;
    for (int e = tid; e < 32768; e += 256) {
        int bb = e >> 12, hw = e & 4095;
        float v = y_pre[((size_t)(bb * 128 + c)) * 4096 + hw];
        s += v; q += v * v;
    }
    ssum[tid] = s; ssq[tid] = q;
    __syncthreads();
    for (int st = 128; st > 0; st >>= 1) {
        if (tid < st) { ssum[tid] += ssum[tid + st]; ssq[tid] += ssq[tid + st]; }
        __syncthreads();
    }
    if (tid == 0) {
        float m = ssum[0] / 32768.f;
        float var = ssq[0] / 32768.f - m * m;
        mu[c] = m;
        rsig[c] = rsqrtf(var + 1e-5f);
    }
}

// ---------------------------------------------------------------------------
// K4: BN + affine + PReLU in place; per (b,c) avg & max pooling
// ---------------------------------------------------------------------------
__global__ __launch_bounds__(256) void k4_bn_prelu_pool(float* __restrict__ y,
                                                        const float* __restrict__ mu,
                                                        const float* __restrict__ rsig,
                                                        const float* __restrict__ gamma,
                                                        const float* __restrict__ beta,
                                                        const float* __restrict__ pa,
                                                        float* __restrict__ avgb,
                                                        float* __restrict__ mxb) {
    __shared__ float ssum[256], smax[256];
    int bc = blockIdx.x, tid = threadIdx.x;
    int c = bc & 127;
    float m = mu[c], rs = rsig[c], g = gamma[c], bt = beta[c], a = pa[c];
    size_t base = (size_t)bc * 4096;
    float s = 0.f, mx = -3.4e38f;
    for (int e = tid; e < 4096; e += 256) {
        float v = (y[base + e] - m) * rs * g + bt;
        v = v > 0.f ? v : a * v;
        y[base + e] = v;
        s += v; mx = fmaxf(mx, v);
    }
    ssum[tid] = s; smax[tid] = mx;
    __syncthreads();
    for (int st = 128; st > 0; st >>= 1) {
        if (tid < st) { ssum[tid] += ssum[tid + st]; smax[tid] = fmaxf(smax[tid], smax[tid + st]); }
        __syncthreads();
    }
    if (tid == 0) { avgb[bc] = ssum[0] / 4096.f; mxb[bc] = smax[0]; }
}

// ---------------------------------------------------------------------------
// K5: channel-attention MLP (128 -> 8 -> 128, shared MLP on avg & max)
// ---------------------------------------------------------------------------
__global__ __launch_bounds__(1024) void k5_ca(const float* __restrict__ avgb,
                                              const float* __restrict__ mxb,
                                              const float* __restrict__ w1,
                                              const float* __restrict__ w2,
                                              float* __restrict__ ca) {
    __shared__ float hA[8][8], hM[8][8];
    int tid = threadIdx.x;
    if (tid < 128) {
        int path = tid >> 6, bb = (tid >> 3) & 7, r = tid & 7;
        const float* v = path ? (mxb + bb * 128) : (avgb + bb * 128);
        float h = 0.f;
        for (int cc = 0; cc < 128; ++cc) h += v[cc] * w1[r * 128 + cc];
        h = h > 0.f ? h : 0.f;
        if (path) hM[bb][r] = h; else hA[bb][r] = h;
    }
    __syncthreads();
    int bb = tid >> 7, cc = tid & 127;
    float o = 0.f;
#pragma unroll
    for (int r = 0; r < 8; ++r) o += (hA[bb][r] + hM[bb][r]) * w2[cc * 8 + r];
    ca[bb * 128 + cc] = 1.f / (1.f + expf(-o));
}

// ---------------------------------------------------------------------------
// K6: apply channel attention (write d_out); channel mean/max for SA input
// ---------------------------------------------------------------------------
__global__ __launch_bounds__(256) void k6_ca_apply(const float* __restrict__ y,
                                                   const float* __restrict__ ca,
                                                   float* __restrict__ out,
                                                   float* __restrict__ smean,
                                                   float* __restrict__ smax) {
    int gid = blockIdx.x * 256 + threadIdx.x;   // 0..32767 = (b,hw)
    int bb = gid >> 12, hw = gid & 4095;
    float s = 0.f, mx = -3.4e38f;
    for (int cc = 0; cc < 128; ++cc) {
        size_t idx = ((size_t)(bb * 128 + cc)) * 4096 + hw;
        float v = y[idx] * ca[bb * 128 + cc];
        out[idx] = v;
        s += v; mx = fmaxf(mx, v);
    }
    smean[gid] = s / 128.f;
    smax[gid]  = mx;
}

// ---------------------------------------------------------------------------
// K7: spatial attention 7x7 conv (2->1 ch, pad 3) + sigmoid + final scale
// ---------------------------------------------------------------------------
__global__ __launch_bounds__(256) void k7_sa(const float* __restrict__ smean,
                                             const float* __restrict__ smax,
                                             const float* __restrict__ w_sa,
                                             const float* __restrict__ b_sa,
                                             float* __restrict__ out) {
    int gid = blockIdx.x * 256 + threadIdx.x;
    int bb = gid >> 12, hw = gid & 4095;
    int h = hw >> 6, w = hw & 63;
    float acc = b_sa[0];
    for (int ty = 0; ty < 7; ++ty) {
        int iy = h + ty - 3;
        if (iy < 0 || iy > 63) continue;
        for (int tx = 0; tx < 7; ++tx) {
            int ix = w + tx - 3;
            if (ix < 0 || ix > 63) continue;
            int sidx = bb * 4096 + iy * 64 + ix;
            acc += smean[sidx] * w_sa[ty * 7 + tx] + smax[sidx] * w_sa[49 + ty * 7 + tx];
        }
    }
    float sa = 1.f / (1.f + expf(-acc));
    for (int cc = 0; cc < 128; ++cc) {
        size_t idx = ((size_t)(bb * 128 + cc)) * 4096 + hw;
        out[idx] *= sa;
    }
}

// ---------------------------------------------------------------------------
extern "C" void kernel_launch(void* const* d_in, const int* in_sizes, int n_in,
                              void* d_out, int out_size, void* d_ws, size_t ws_size,
                              hipStream_t stream) {
    (void)in_sizes; (void)n_in; (void)out_size; (void)ws_size;
    const float* x       = (const float*)d_in[0];
    const float* w_off   = (const float*)d_in[1];
    const float* b_off   = (const float*)d_in[2];
    const float* w_dcn   = (const float*)d_in[3];
    const float* b_dcn   = (const float*)d_in[4];
    const float* gamma   = (const float*)d_in[5];
    const float* beta    = (const float*)d_in[6];
    const float* prelu_a = (const float*)d_in[7];
    const float* w1      = (const float*)d_in[8];
    const float* w2      = (const float*)d_in[9];
    const float* w_sa    = (const float*)d_in[10];
    const float* b_sa    = (const float*)d_in[11];
    float* out = (float*)d_out;

    // workspace carve-out (~21.0 MB)
    char* ws = (char*)d_ws;
    size_t off = 0;
    auto carve = [&](size_t bytes) -> void* {
        void* p = ws + off;
        off = (off + bytes + 255) & ~(size_t)255;
        return p;
    };
    __bf16* wA   = (__bf16*)carve(147456 * 2);          // 128 x 1152 bf16
    __bf16* wOff = (__bf16*)carve(36864 * 2);           // 32  x 1152 bf16 (padded)
    float* ysb   = (float*)carve(8 * 9 * 4096 * 4);
    float* xsb   = (float*)carve(8 * 9 * 4096 * 4);
    float* mb    = (float*)carve(8 * 9 * 4096 * 4);
    float* y_pre = (float*)carve((size_t)8 * 128 * 4096 * 4);
    float* mu    = (float*)carve(128 * 4);
    float* rsig  = (float*)carve(128 * 4);
    float* avgb  = (float*)carve(1024 * 4);
    float* mxb   = (float*)carve(1024 * 4);
    float* ca    = (float*)carve(1024 * 4);
    float* smean = (float*)carve(32768 * 4);
    float* smax  = (float*)carve(32768 * 4);

    k0_convert_wdcn<<<(147456 + 255) / 256, 256, 0, stream>>>(w_dcn, wA, 147456);
    k0_convert_woff<<<(36864 + 255) / 256, 256, 0, stream>>>(w_off, wOff);
    k1_offset_wmma<<<256, 256, 0, stream>>>(x, wOff, b_off, ysb, xsb, mb);
    k2_dcn_wmma<<<256, 256, 0, stream>>>(x, wA, ysb, xsb, mb, b_dcn, y_pre);
    k3_bn_stats<<<128, 256, 0, stream>>>(y_pre, mu, rsig);
    k4_bn_prelu_pool<<<1024, 256, 0, stream>>>(y_pre, mu, rsig, gamma, beta, prelu_a, avgb, mxb);
    k5_ca<<<1, 1024, 0, stream>>>(avgb, mxb, w1, w2, ca);
    k6_ca_apply<<<128, 256, 0, stream>>>(y_pre, ca, out, smean, smax);
    k7_sa<<<128, 256, 0, stream>>>(smean, smax, w_sa, b_sa, out);
}
